// DeepSeekV2ForCausalLM_87393994539793
// MI455X (gfx1250) — matrix operand order repaired and checked by reading, not compiled, not run on hardware
//
#include <hip/hip_runtime.h>
#include <hip/hip_bf16.h>

// ---------- problem constants ----------
#define T_DIM 2048
#define S_DIM 2048
#define H_DIM 16
#define DN    128
#define DR    64
#define DV    128
#define CLR   512
#define ODIM  2048
#define KD    192                         // DN + DR
#define ATT_SCALE 0.07216878364870323f    // 1/sqrt(192)
#define NEGINF (-1.0e30f)

// ---------- WMMA types / helpers ----------
typedef __attribute__((ext_vector_type(16))) __bf16 v16bf;
typedef __attribute__((ext_vector_type(8)))  __bf16 v8bf;
typedef __attribute__((ext_vector_type(8)))  float  v8f;
typedef __attribute__((ext_vector_type(4)))  unsigned int v4u;
typedef __attribute__((ext_vector_type(8)))  int v8i;
typedef __attribute__((ext_vector_type(4)))  int v4i;

__device__ __forceinline__ unsigned short f2bf(float f) {
  unsigned int u = __builtin_bit_cast(unsigned int, f);
  unsigned int r = u + 0x7FFFu + ((u >> 16) & 1u);   // round-to-nearest-even
  return (unsigned short)(r >> 16);
}

// Build a v16bf fragment from two contiguous 16-byte chunks (LDS or global).
__device__ __forceinline__ v16bf load_frag(const unsigned short* p0,
                                           const unsigned short* p1) {
  v8bf lo = *(const v8bf*)p0;
  v8bf hi = *(const v8bf*)p1;
  return __builtin_shufflevector(lo, hi, 0,1,2,3,4,5,6,7,8,9,10,11,12,13,14,15);
}

__device__ __forceinline__ v8f wmma_bf16(v16bf a, v16bf b, v8f c) {
  // D = A(16x32) * B(32x16) + C, f32 accumulate
  return __builtin_amdgcn_wmma_f32_16x16x32_bf16(false, a, false, b,
                                                 (short)0, c, false, false);
}

// Butterfly reductions over 16-lane halves (rows i / i+8 stay separate).
__device__ __forceinline__ float rmax16(float v) {
#pragma unroll
  for (int m = 1; m < 16; m <<= 1) v = fmaxf(v, __shfl_xor(v, m, 32));
  return v;
}
__device__ __forceinline__ float rsum16(float v) {
#pragma unroll
  for (int m = 1; m < 16; m <<= 1) v += __shfl_xor(v, m, 32);
  return v;
}

#if __has_builtin(__builtin_amdgcn_tensor_load_to_lds) && \
    __has_builtin(__builtin_amdgcn_s_wait_tensorcnt)
#define USE_TDM 1
#else
#define USE_TDM 0
#endif

#if USE_TDM
// Issue a 2-D TDM tile load (bf16 elements): tile_d0 x tile_d1 tile whose rows
// are tile_d0 elements, row stride stride0 elements, into contiguous LDS.
// D# bitfields per CDNA5 ISA ch.8 (group0 128b, group1 256b; groups 2/3 unused).
// This toolchain exposes the 6-arg builtin:
//   (uint32x4 g0, int32x8 g1, int32x4 g2, int32x4 g3, int32x8 g4, i32 cpol)
__device__ __forceinline__ void tdm_load_2d(unsigned lds_addr,
                                            unsigned long long gaddr,
                                            unsigned tile_d0, unsigned tile_d1,
                                            unsigned long long stride0) {
  v4u g0;
  g0[0] = 1u;                                                   // count=1, user D#
  g0[1] = lds_addr;                                             // lds_addr (bytes)
  g0[2] = (unsigned)(gaddr & 0xFFFFFFFFull);                    // global_addr[31:0]
  g0[3] = (unsigned)((gaddr >> 32) & 0x01FFFFFFull) | (2u << 30); // addr[56:32] | type=2
  v8i g1;
  g1[0] = (int)(1u << 16);                                      // data_size=1 (2B); wg_mask=0
  g1[1] = (int)((tile_d0 & 0xFFFFu) << 16);                     // tensor_dim0[15:0]
  g1[2] = (int)(((tile_d0 >> 16) & 0xFFFFu) |
                ((tile_d1 & 0xFFFFu) << 16));                   // tensor_dim0[31:16]|tensor_dim1[15:0]
  g1[3] = (int)(((tile_d1 >> 16) & 0xFFFFu) |
                ((tile_d0 & 0xFFFFu) << 16));                   // tensor_dim1[31:16]|tile_dim0
  g1[4] = (int)(tile_d1 & 0xFFFFu);                             // tile_dim1; tile_dim2=0
  g1[5] = (int)(unsigned)(stride0 & 0xFFFFFFFFull);             // tensor_dim0_stride[31:0]
  g1[6] = (int)(unsigned)((stride0 >> 32) & 0xFFFFull);         // stride0[47:32]; stride1 lo=0
  g1[7] = 0;                                                    // stride1 hi = 0
  v4i gz4 = (v4i){0, 0, 0, 0};
  v8i gz8 = (v8i){0, 0, 0, 0, 0, 0, 0, 0};
  __builtin_amdgcn_tensor_load_to_lds(g0, g1, gz4, gz4, gz8, 0);
}
__device__ __forceinline__ unsigned lds_off(const void* p) {
  return (unsigned)(unsigned long long)(size_t)p;   // flat->LDS: low 32 bits
}
#endif

// =====================================================================
// Kernel 0: copy rope-K (pe_all) into KNPE[h][s][128..192) as bf16
// =====================================================================
__global__ __launch_bounds__(256) void pefill_kernel(
    const float* __restrict__ pe, unsigned short* __restrict__ knpe) {
  int idx = blockIdx.x * 256 + threadIdx.x;          // H*S*DR = 2,097,152
  if (idx < H_DIM * S_DIM * DR) {
    int r = idx & (DR - 1);
    int s = (idx >> 6) & (S_DIM - 1);
    int h = idx >> 17;
    knpe[((size_t)h * S_DIM + s) * KD + DN + r] = f2bf(pe[(size_t)s * DR + r]);
  }
}

// =====================================================================
// Kernel 1: k/v decompression GEMM.
//   out[s, e] = sum_c kv[s,c] * wkv_b[h, ct*128+e, c]
//   ct==0 -> k_nope rows of KNPE (bf16);  ct==1 -> Vt[h][d][s] (bf16)
// =====================================================================
__global__ __launch_bounds__(256) void kvproj_kernel(
    const float* __restrict__ kv, const float* __restrict__ wkv,
    unsigned short* __restrict__ knpe, unsigned short* __restrict__ vt) {
  const int h  = blockIdx.z;
  const int ct = blockIdx.y;
  const int sb = blockIdx.x * 128;
  const int tid = threadIdx.x;
  const int wave = tid >> 5, lane = tid & 31;
  const int wr = wave >> 2, wc = wave & 3;
  const int lm = lane & 15, lh = lane >> 4;

  __shared__ __align__(16) unsigned short As[128 * 32];  // s x c (row major)
  __shared__ __align__(16) unsigned short Bs[128 * 32];  // e x c (B transposed)

  v8f acc[4][2];
#pragma unroll
  for (int mi = 0; mi < 4; ++mi)
#pragma unroll
    for (int nj = 0; nj < 2; ++nj)
      acc[mi][nj] = (v8f){0.f, 0.f, 0.f, 0.f, 0.f, 0.f, 0.f, 0.f};

  for (int k0 = 0; k0 < CLR; k0 += 32) {
    __syncthreads();
    for (int idx = tid * 4; idx < 128 * 32; idx += 1024) {
      int r = idx >> 5, c = idx & 31;
      float4 f = *(const float4*)(kv + (size_t)(sb + r) * CLR + k0 + c);
      As[idx + 0] = f2bf(f.x); As[idx + 1] = f2bf(f.y);
      As[idx + 2] = f2bf(f.z); As[idx + 3] = f2bf(f.w);
    }
    for (int idx = tid * 4; idx < 128 * 32; idx += 1024) {
      int r = idx >> 5, c = idx & 31;
      float4 f = *(const float4*)(wkv + ((size_t)h * 256 + ct * 128 + r) * CLR + k0 + c);
      Bs[idx + 0] = f2bf(f.x); Bs[idx + 1] = f2bf(f.y);
      Bs[idx + 2] = f2bf(f.z); Bs[idx + 3] = f2bf(f.w);
    }
    __syncthreads();

    v16bf af[4], bfr[2];
#pragma unroll
    for (int mi = 0; mi < 4; ++mi)
      af[mi] = load_frag(As + (wr * 64 + mi * 16 + lm) * 32 + lh * 16,
                         As + (wr * 64 + mi * 16 + lm) * 32 + lh * 16 + 8);
#pragma unroll
    for (int nj = 0; nj < 2; ++nj)
      bfr[nj] = load_frag(Bs + (wc * 32 + nj * 16 + lm) * 32 + lh * 16,
                          Bs + (wc * 32 + nj * 16 + lm) * 32 + lh * 16 + 8);
#pragma unroll
    for (int mi = 0; mi < 4; ++mi)
#pragma unroll
      for (int nj = 0; nj < 2; ++nj)
        acc[mi][nj] = wmma_bf16(af[mi], bfr[nj], acc[mi][nj]);
  }

#pragma unroll
  for (int mi = 0; mi < 4; ++mi)
#pragma unroll
    for (int nj = 0; nj < 2; ++nj)
#pragma unroll
      for (int i = 0; i < 8; ++i) {
        int s   = sb + wr * 64 + mi * 16 + i + 8 * lh;
        int col = wc * 32 + nj * 16 + lm;
        unsigned short b = f2bf(acc[mi][nj][i]);
        if (ct == 0) knpe[((size_t)h * S_DIM + s) * KD + col] = b;      // k_nope
        else         vt[((size_t)h * DV + col) * S_DIM + s]   = b;      // V^T
      }
}

// =====================================================================
// Kernel 2: flash attention, TDM double-buffered K/V staging.
// Block: 128 thr (4 waves), grid (T/64, H).
// =====================================================================
__global__ __launch_bounds__(128) void attn_kernel(
    const float* __restrict__ qn, const float* __restrict__ qpe,
    const unsigned short* __restrict__ knpe, const unsigned short* __restrict__ vt,
    const int* __restrict__ pcl_ptr, float* __restrict__ ov) {
  const int h    = blockIdx.y;
  const int tblk = blockIdx.x;
  const int tid  = threadIdx.x;
  const int wave = tid >> 5, lane = tid & 31;
  const int lm = lane & 15, lh = lane >> 4;
  const int t0 = tblk * 64 + wave * 16;
  const int pcl = pcl_ptr[0];

  __shared__ __align__(16) unsigned short Ks[2][32 * KD];   // 24 KB double buffer
  __shared__ __align__(16) unsigned short Vs[2][128 * 32];  // 16 KB double buffer
  __shared__ __align__(16) unsigned short Ps[4][16 * 32];   //  4 KB per-wave P

  // ---- Q fragments (A-layout, 6 K-steps of 32 over [q_nope | q_pe]) ----
  v16bf qf[6];
  {
    const int trow = t0 + lm;
#pragma unroll
    for (int kk = 0; kk < 6; ++kk) {
      v16bf f;
#pragma unroll
      for (int e = 0; e < 16; ++e) {
        int kloc = lh * 16 + e;
        int k = kk * 32 + kloc;
        float val = (k < DN)
            ? qn[((size_t)trow * H_DIM + h) * DN + k]
            : qpe[((size_t)trow * H_DIM + h) * DR + (k - DN)];
        f[e] = __builtin_bit_cast(__bf16, f2bf(val));
      }
      qf[kk] = f;
    }
  }

  v8f o[8];
#pragma unroll
  for (int nj = 0; nj < 8; ++nj)
    o[nj] = (v8f){0.f, 0.f, 0.f, 0.f, 0.f, 0.f, 0.f, 0.f};
  float m_run[8], l_run[8];
#pragma unroll
  for (int i = 0; i < 8; ++i) { m_run[i] = NEGINF; l_run[i] = 0.f; }

  const unsigned short* kbase = knpe + (size_t)h * S_DIM * KD;
  const unsigned short* vbase = vt + (size_t)h * DV * S_DIM;

  // causal: need keys s <= t_max + pcl;  t_max = tblk*64 + 63
  int nsb = (tblk * 64 + 64 + pcl + 31) >> 5;
  if (nsb > S_DIM / 32) nsb = S_DIM / 32;

#if USE_TDM
  // prologue: DMA block 0 into buffer 0
  if (wave == 0) {
    tdm_load_2d(lds_off(Ks[0]), (unsigned long long)(size_t)kbase, KD, 32, KD);
    tdm_load_2d(lds_off(Vs[0]), (unsigned long long)(size_t)vbase, 32, 128, S_DIM);
  }
#endif

  for (int sb = 0; sb < nsb; ++sb) {
    const int s0 = sb * 32;
    const int buf = sb & 1;
#if USE_TDM
    if (wave == 0) {
      if (sb + 1 < nsb) {   // issue next block's DMA, then wait current done
        const int s1 = s0 + 32;
        tdm_load_2d(lds_off(Ks[buf ^ 1]),
                    (unsigned long long)(size_t)(kbase + (size_t)s1 * KD),
                    KD, 32, KD);
        tdm_load_2d(lds_off(Vs[buf ^ 1]),
                    (unsigned long long)(size_t)(vbase + s1),
                    32, 128, S_DIM);
        __builtin_amdgcn_s_wait_tensorcnt(2);
      } else {
        __builtin_amdgcn_s_wait_tensorcnt(0);
      }
    }
    __syncthreads();
#else
    __syncthreads();
    for (int idx = tid * 8; idx < 32 * KD; idx += 1024) {
      int r = idx / KD, c = idx % KD;
      *(uint4*)(Ks[buf] + idx) = *(const uint4*)(kbase + (size_t)(s0 + r) * KD + c);
    }
    for (int idx = tid * 8; idx < 128 * 32; idx += 1024) {
      int r = idx >> 5, c = idx & 31;
      *(uint4*)(Vs[buf] + idx) = *(const uint4*)(vbase + (size_t)r * S_DIM + s0 + c);
    }
    __syncthreads();
#endif

    // ---- logits = Q (16x192) @ K^T (192x32): 2 n-tiles x 6 k-steps ----
    v8f lg[2];
    lg[0] = (v8f){0.f, 0.f, 0.f, 0.f, 0.f, 0.f, 0.f, 0.f};
    lg[1] = lg[0];
#pragma unroll
    for (int kk = 0; kk < 6; ++kk) {
#pragma unroll
      for (int nj = 0; nj < 2; ++nj) {
        const unsigned short* bp = Ks[buf] + (nj * 16 + lm) * KD + kk * 32 + lh * 16;
        lg[nj] = wmma_bf16(qf[kk], load_frag(bp, bp + 8), lg[nj]);
      }
    }

    // ---- scale, causal mask, online softmax ----
#pragma unroll
    for (int i = 0; i < 8; ++i) {
      int t = t0 + i + 8 * lh;
#pragma unroll
      for (int nj = 0; nj < 2; ++nj) {
        int s = s0 + nj * 16 + lm;
        float v = lg[nj][i] * ATT_SCALE;
        if (s > t + pcl) v = NEGINF;
        lg[nj][i] = v;
      }
      float bmax = rmax16(fmaxf(lg[0][i], lg[1][i]));
      float mn = fmaxf(m_run[i], bmax);
      float al = __expf(m_run[i] - mn);
      m_run[i] = mn;
      float p0 = __expf(lg[0][i] - mn);
      float p1 = __expf(lg[1][i] - mn);
      lg[0][i] = p0; lg[1][i] = p1;
      l_run[i] = l_run[i] * al + rsum16(p0 + p1);
#pragma unroll
      for (int nj = 0; nj < 8; ++nj) o[nj][i] = o[nj][i] * al;
    }

    // ---- P: C-layout -> A-layout via per-wave LDS bounce (bf16) ----
#pragma unroll
    for (int nj = 0; nj < 2; ++nj)
#pragma unroll
      for (int i = 0; i < 8; ++i)
        Ps[wave][(i + 8 * lh) * 32 + nj * 16 + lm] = f2bf(lg[nj][i]);
    asm volatile("s_wait_dscnt 0" ::: "memory");   // DS in-order within wave
    v16bf pf = load_frag(Ps[wave] + lm * 32 + lh * 16,
                         Ps[wave] + lm * 32 + lh * 16 + 8);

    // ---- O += P (16x32) @ V (32x128): 8 n-tiles over dv ----
#pragma unroll
    for (int nj = 0; nj < 8; ++nj) {
      const unsigned short* bp = Vs[buf] + (nj * 16 + lm) * 32 + lh * 16;
      o[nj] = wmma_bf16(pf, load_frag(bp, bp + 8), o[nj]);
    }
    __syncthreads();   // all waves done reading buf before it is refilled
  }

  // ---- normalize and write out_v (T, H*DV) f32 ----
#pragma unroll
  for (int i = 0; i < 8; ++i) {
    float inv = 1.f / l_run[i];
    int t = t0 + i + 8 * lh;
#pragma unroll
    for (int nj = 0; nj < 8; ++nj)
      ov[(size_t)t * (H_DIM * DV) + h * DV + nj * 16 + lm] = o[nj][i] * inv;
  }
}

// =====================================================================
// Kernel 3: out = ov (2048x2048) @ wo^T (2048x2048), f32 out, bf16 WMMA.
// =====================================================================
__global__ __launch_bounds__(256) void outproj_kernel(
    const float* __restrict__ ovg, const float* __restrict__ wo,
    float* __restrict__ out) {
  const int tb = blockIdx.x * 128;
  const int ob = blockIdx.y * 128;
  const int tid = threadIdx.x;
  const int wave = tid >> 5, lane = tid & 31;
  const int wr = wave >> 2, wc = wave & 3;
  const int lm = lane & 15, lh = lane >> 4;

  __shared__ __align__(16) unsigned short As[128 * 32];  // t x e
  __shared__ __align__(16) unsigned short Bs[128 * 32];  // o x e (B transposed)

  v8f acc[4][2];
#pragma unroll
  for (int mi = 0; mi < 4; ++mi)
#pragma unroll
    for (int nj = 0; nj < 2; ++nj)
      acc[mi][nj] = (v8f){0.f, 0.f, 0.f, 0.f, 0.f, 0.f, 0.f, 0.f};

  for (int k0 = 0; k0 < ODIM; k0 += 32) {
    __syncthreads();
    for (int idx = tid * 4; idx < 128 * 32; idx += 1024) {
      int r = idx >> 5, c = idx & 31;
      float4 f = *(const float4*)(ovg + (size_t)(tb + r) * ODIM + k0 + c);
      As[idx + 0] = f2bf(f.x); As[idx + 1] = f2bf(f.y);
      As[idx + 2] = f2bf(f.z); As[idx + 3] = f2bf(f.w);
    }
    for (int idx = tid * 4; idx < 128 * 32; idx += 1024) {
      int r = idx >> 5, c = idx & 31;
      float4 f = *(const float4*)(wo + (size_t)(ob + r) * ODIM + k0 + c);
      Bs[idx + 0] = f2bf(f.x); Bs[idx + 1] = f2bf(f.y);
      Bs[idx + 2] = f2bf(f.z); Bs[idx + 3] = f2bf(f.w);
    }
    __syncthreads();

    v16bf af[4], bfr[2];
#pragma unroll
    for (int mi = 0; mi < 4; ++mi)
      af[mi] = load_frag(As + (wr * 64 + mi * 16 + lm) * 32 + lh * 16,
                         As + (wr * 64 + mi * 16 + lm) * 32 + lh * 16 + 8);
#pragma unroll
    for (int nj = 0; nj < 2; ++nj)
      bfr[nj] = load_frag(Bs + (wc * 32 + nj * 16 + lm) * 32 + lh * 16,
                          Bs + (wc * 32 + nj * 16 + lm) * 32 + lh * 16 + 8);
#pragma unroll
    for (int mi = 0; mi < 4; ++mi)
#pragma unroll
      for (int nj = 0; nj < 2; ++nj)
        acc[mi][nj] = wmma_bf16(af[mi], bfr[nj], acc[mi][nj]);
  }

#pragma unroll
  for (int mi = 0; mi < 4; ++mi)
#pragma unroll
    for (int nj = 0; nj < 2; ++nj)
#pragma unroll
      for (int i = 0; i < 8; ++i) {
        int t = tb + wr * 64 + mi * 16 + i + 8 * lh;
        int c = ob + wc * 32 + nj * 16 + lm;
        out[(size_t)t * ODIM + c] = acc[mi][nj][i];
      }
}

// =====================================================================
// host launcher
// =====================================================================
extern "C" void kernel_launch(void* const* d_in, const int* in_sizes, int n_in,
                              void* d_out, int out_size, void* d_ws, size_t ws_size,
                              hipStream_t stream) {
  const float* qn  = (const float*)d_in[0];   // (T,H,DN)
  const float* qpe = (const float*)d_in[1];   // (T,H,DR)
  const float* kv  = (const float*)d_in[2];   // (S,CLR)
  const float* pe  = (const float*)d_in[3];   // (S,DR)
  const float* wkv = (const float*)d_in[4];   // (H,256,CLR)
  const float* wo  = (const float*)d_in[5];   // (ODIM,H*DV)
  const int*   pcl = (const int*)d_in[6];
  float* out = (float*)d_out;

  char* ws = (char*)d_ws;
  unsigned short* knpe = (unsigned short*)ws;                      // H*S*192 bf16 = 12.6 MB
  unsigned short* vt   = (unsigned short*)(ws + 12582912);         // H*DV*S bf16 =  8.4 MB
  float*          ov   = (float*)(ws + 20971520);                  // T*H*DV f32  = 16.8 MB

  pefill_kernel<<<(H_DIM * S_DIM * DR + 255) / 256, 256, 0, stream>>>(pe, knpe);

  dim3 gA(S_DIM / 128, 2, H_DIM);
  kvproj_kernel<<<gA, 256, 0, stream>>>(kv, wkv, knpe, vt);

  dim3 gB(T_DIM / 64, H_DIM);
  attn_kernel<<<gB, 128, 0, stream>>>(qn, qpe, knpe, vt, pcl, ov);

  dim3 gC(T_DIM / 128, ODIM / 128);
  outproj_kernel<<<gC, 256, 0, stream>>>(ov, wo, out);
}
